// MultiHeadGraphAttention_70179765616692
// MI455X (gfx1250) — compile-verified, matrix operand order checked by reading, not run on hardware
//
#include <hip/hip_runtime.h>

#define N_NODES 6144
#define F_IN    64
#define F_OUT   32
#define N_HEAD  4
#define TJ      64          // j-columns per LDS stage (2 WMMA K-steps)
#define HPT_S   72          // hpT row stride (ushorts): 144B = 36 dwords, b128-aligned rows

typedef __attribute__((ext_vector_type(16))) __bf16 v16bf;
typedef __attribute__((ext_vector_type(8)))  float  v8f;

union BF16x16 {
  v16bf v;
  unsigned int w[8];
  uint4 q[2];
};

// round-to-nearest-even fp32 -> bf16 bits (cold path only: tile staging / init)
__device__ __forceinline__ unsigned int f32_to_bf16(float f) {
  unsigned int x = __float_as_uint(f);
  x += 0x7FFFu + ((x >> 16) & 1u);
  return x >> 16;
}
__device__ __forceinline__ unsigned int pack2bf(float lo, float hi) {
  return f32_to_bf16(lo) | (f32_to_bf16(hi) << 16);
}
__device__ __forceinline__ unsigned int dup_bf16(float f) {
  unsigned int b = f32_to_bf16(f);
  return b | (b << 16);
}

// packed bf16 ops (CDNA5 VOP3P: V_PK_MUL_BF16 / V_PK_MAX_NUM_BF16)
__device__ __forceinline__ unsigned int pk_mul_bf16(unsigned int a, unsigned int b) {
  unsigned int d;
  asm("v_pk_mul_bf16 %0, %1, %2" : "=v"(d) : "v"(a), "v"(b));
  return d;
}
__device__ __forceinline__ unsigned int pk_max_bf16(unsigned int a, unsigned int b) {
  unsigned int d;
  asm("v_pk_max_num_bf16 %0, %1, %2" : "=v"(d) : "v"(a), "v"(b));
  return d;
}

// ---------------------------------------------------------------------------
// Kernel 1: h_prime = h @ w[head] (fp32 FMA; 0.1 GFLOP) + s,d projections.
// ---------------------------------------------------------------------------
__global__ __launch_bounds__(256) void gat_prep(
    const float* __restrict__ h, const float* __restrict__ w,
    const float* __restrict__ a_src, const float* __restrict__ a_dst,
    unsigned short* __restrict__ hp, float* __restrict__ sArr,
    float* __restrict__ dArr) {
  const int head = blockIdx.y;
  const int wave = threadIdx.x >> 5;
  const int lane = threadIdx.x & 31;
  const int row  = blockIdx.x * 8 + wave;

  const float* hRow = h + (size_t)row * F_IN;
  const float* wH   = w + (size_t)head * F_IN * F_OUT + lane;

  float acc = 0.f;
#pragma unroll 8
  for (int k = 0; k < F_IN; ++k)
    acc = fmaf(hRow[k], wH[k * F_OUT], acc);

  hp[((size_t)head * N_NODES + row) * F_OUT + lane] =
      (unsigned short)f32_to_bf16(acc);

  float sv = acc * a_src[head * F_OUT + lane];
  float dv = acc * a_dst[head * F_OUT + lane];
#pragma unroll
  for (int m = 16; m >= 1; m >>= 1) {
    sv += __shfl_xor(sv, m, 32);
    dv += __shfl_xor(dv, m, 32);
  }
  if (lane == 0) {
    sArr[head * N_NODES + row] = sv;
    dArr[head * N_NODES + row] = dv;
  }
}

// ---------------------------------------------------------------------------
// Kernel 2: dmax[head] = max_j d[head][j]
// ---------------------------------------------------------------------------
__global__ __launch_bounds__(256) void gat_dmax(const float* __restrict__ dArr,
                                                float* __restrict__ dmax) {
  __shared__ float red[256];
  const int head = blockIdx.x;
  float m = -3.402823466e38f;
  for (int i = threadIdx.x; i < N_NODES; i += 256)
    m = fmaxf(m, dArr[head * N_NODES + i]);
  red[threadIdx.x] = m;
  __syncthreads();
  for (int s = 128; s > 0; s >>= 1) {
    if (threadIdx.x < s) red[threadIdx.x] = fmaxf(red[threadIdx.x], red[threadIdx.x + s]);
    __syncthreads();
  }
  if (threadIdx.x == 0) dmax[head] = red[0];
}

// ---------------------------------------------------------------------------
// Kernel 3: softmax(leaky_relu(s_i+d_j)) @ V, bf16 WMMA + packed-bf16 VALU.
//   p = exp(lrelu(s+d)-m) = max(e1*exp(d-dm), e2*exp(0.2(d-dm)))
// computed entirely as v_pk_mul_bf16 / v_pk_max_num_bf16 on bf16 pairs.
// Row sums Z via an extra WMMA against an all-ones B fragment (lands in the
// same lane/VGPR slot as the outputs -> no shuffles, no scalar adds).
// Wave owns 32 rows; 64-column LDS stages, software-pipelined global->LDS.
// ---------------------------------------------------------------------------
__global__ __launch_bounds__(128) void gat_attn(
    const float* __restrict__ sArr, const float* __restrict__ dArr,
    const unsigned short* __restrict__ hp, const float* __restrict__ dmax,
    const float* __restrict__ bias, float* __restrict__ out) {
  __shared__ alignas(16) unsigned short hpT[F_OUT][HPT_S]; // V transposed [feat][j]
  __shared__ alignas(16) uint2 f12pk[TJ / 2];  // {bf16x2 f1, bf16x2 f2} per j-pair

  const int head = blockIdx.y;
  const int tid  = threadIdx.x;
  const int wave = tid >> 5;
  const int lane = tid & 31;
  const int hi   = lane >> 4;
  const int mloc = lane & 15;
  const int rowBase = blockIdx.x * 128 + wave * 32;

  const unsigned short* hpH = hp + (size_t)head * N_NODES * F_OUT;
  const float dm = dmax[head];

  // per-lane row factors (two 16-row groups), broadcast as bf16 pairs
  const float s0 = sArr[head * N_NODES + rowBase + mloc];
  const float s1 = sArr[head * N_NODES + rowBase + 16 + mloc];
  const float x0 = s0 + dm, x1 = s1 + dm;
  const float m0 = x0 > 0.f ? x0 : 0.2f * x0;
  const float m1 = x1 > 0.f ? x1 : 0.2f * x1;
  const unsigned int e1_0p = dup_bf16(__expf(x0 - m0));
  const unsigned int e2_0p = dup_bf16(__expf(0.2f * x0 - m0));
  const unsigned int e1_1p = dup_bf16(__expf(x1 - m1));
  const unsigned int e2_1p = dup_bf16(__expf(0.2f * x1 - m1));

  BF16x16 ones;
#pragma unroll
  for (int r = 0; r < 8; ++r) ones.w[r] = 0x3F803F80u;  // bf16(1.0) x2

  v8f c00 = {}, c01 = {}, c10 = {}, c11 = {}, cz0 = {}, cz1 = {};

  // staging: thread owns j-pair (tid>>2)*2, feature quad (tid&3)*8
  const int jp = (tid >> 2) * 2;       // even j within tile
  const int fq = (tid & 3) * 8;        // feature base
  uint4 pg0 = *(const uint4*)(hpH + (size_t)jp * F_OUT + fq);
  uint4 pg1 = *(const uint4*)(hpH + (size_t)(jp + 1) * F_OUT + fq);
  float2 pdv = make_float2(0.f, 0.f);
  if (tid < TJ / 2) pdv = *(const float2*)(dArr + head * N_NODES + 2 * tid);

  for (int jt = 0; jt < N_NODES; jt += TJ) {
    __syncthreads();  // previous stage fully consumed
    {
      // transposed store: interleave the two j-rows -> b32 stores
      unsigned int lo, hv;
      lo = (pg0.x & 0xFFFFu) | (pg1.x << 16);
      hv = (pg0.x >> 16) | (pg1.x & 0xFFFF0000u);
      *(unsigned int*)&hpT[fq + 0][jp] = lo;
      *(unsigned int*)&hpT[fq + 1][jp] = hv;
      lo = (pg0.y & 0xFFFFu) | (pg1.y << 16);
      hv = (pg0.y >> 16) | (pg1.y & 0xFFFF0000u);
      *(unsigned int*)&hpT[fq + 2][jp] = lo;
      *(unsigned int*)&hpT[fq + 3][jp] = hv;
      lo = (pg0.z & 0xFFFFu) | (pg1.z << 16);
      hv = (pg0.z >> 16) | (pg1.z & 0xFFFF0000u);
      *(unsigned int*)&hpT[fq + 4][jp] = lo;
      *(unsigned int*)&hpT[fq + 5][jp] = hv;
      lo = (pg0.w & 0xFFFFu) | (pg1.w << 16);
      hv = (pg0.w >> 16) | (pg1.w & 0xFFFF0000u);
      *(unsigned int*)&hpT[fq + 6][jp] = lo;
      *(unsigned int*)&hpT[fq + 7][jp] = hv;
      if (tid < TJ / 2) {
        const float t0 = pdv.x - dm, t1 = pdv.y - dm;
        f12pk[tid] = make_uint2(pack2bf(__expf(t0), __expf(t1)),
                                pack2bf(__expf(0.2f * t0), __expf(0.2f * t1)));
      }
    }
    // prefetch next stage while this one computes
    const int jn = jt + TJ;
    if (jn < N_NODES) {
      pg0 = *(const uint4*)(hpH + (size_t)(jn + jp) * F_OUT + fq);
      pg1 = *(const uint4*)(hpH + (size_t)(jn + jp + 1) * F_OUT + fq);
      if (tid < TJ / 2) pdv = *(const float2*)(dArr + head * N_NODES + jn + 2 * tid);
    }
    __syncthreads();  // stage visible

#pragma unroll
    for (int kk = 0; kk < 2; ++kk) {
      const int kb = kk * 32;
      BF16x16 a0, a1, b0, b1;
      // B fragments: 2x ds_load_b128 each (rows conflict-free, stride 36 dwords)
      b0.q[0] = *(const uint4*)&hpT[mloc][kb + 8 * hi];
      b0.q[1] = *(const uint4*)&hpT[mloc][kb + 16 + 8 * hi];
      b1.q[0] = *(const uint4*)&hpT[16 + mloc][kb + 8 * hi];
      b1.q[1] = *(const uint4*)&hpT[16 + mloc][kb + 16 + 8 * hi];
      // f pairs for this lane-group's K set: 4x ds_load_b128
      const uint4 fA = *(const uint4*)&f12pk[kb / 2 + 4 * hi];      // r=0,1
      const uint4 fB = *(const uint4*)&f12pk[kb / 2 + 4 * hi + 2];  // r=2,3
      const uint4 fC = *(const uint4*)&f12pk[kb / 2 + 8 + 4 * hi];      // r=4,5
      const uint4 fD = *(const uint4*)&f12pk[kb / 2 + 8 + 4 * hi + 2];  // r=6,7
#define PFRAG(idx, F1, F2)                                              \
      a0.w[idx] = pk_max_bf16(pk_mul_bf16(e1_0p, F1), pk_mul_bf16(e2_0p, F2)); \
      a1.w[idx] = pk_max_bf16(pk_mul_bf16(e1_1p, F1), pk_mul_bf16(e2_1p, F2));
      PFRAG(0, fA.x, fA.y)
      PFRAG(1, fA.z, fA.w)
      PFRAG(2, fB.x, fB.y)
      PFRAG(3, fB.z, fB.w)
      PFRAG(4, fC.x, fC.y)
      PFRAG(5, fC.z, fC.w)
      PFRAG(6, fD.x, fD.y)
      PFRAG(7, fD.z, fD.w)
#undef PFRAG
      c00 = __builtin_amdgcn_wmma_f32_16x16x32_bf16(false, a0.v, false, b0.v,
                                                    (short)0, c00, false, false);
      c01 = __builtin_amdgcn_wmma_f32_16x16x32_bf16(false, a0.v, false, b1.v,
                                                    (short)0, c01, false, false);
      c10 = __builtin_amdgcn_wmma_f32_16x16x32_bf16(false, a1.v, false, b0.v,
                                                    (short)0, c10, false, false);
      c11 = __builtin_amdgcn_wmma_f32_16x16x32_bf16(false, a1.v, false, b1.v,
                                                    (short)0, c11, false, false);
      cz0 = __builtin_amdgcn_wmma_f32_16x16x32_bf16(false, a0.v, false, ones.v,
                                                    (short)0, cz0, false, false);
      cz1 = __builtin_amdgcn_wmma_f32_16x16x32_bf16(false, a1.v, false, ones.v,
                                                    (short)0, cz1, false, false);
    }
  }

  // cz[r] = Z of row (r + 8*hi) of each group, already in the right lane
  const float bias0 = bias[mloc];
  const float bias1 = bias[16 + mloc];
  const int moff = hi ? 8 : 0;
#pragma unroll
  for (int r = 0; r < 8; ++r) {
    const int rowM = moff + r;
    const float inv0 = 1.0f / cz0[r];
    const float inv1 = 1.0f / cz1[r];
    float* o0 = out + (size_t)(rowBase + rowM) * (N_HEAD * F_OUT) + head * F_OUT;
    float* o1 = out + (size_t)(rowBase + 16 + rowM) * (N_HEAD * F_OUT) + head * F_OUT;
    o0[mloc]      = c00[r] * inv0 + bias0;
    o0[16 + mloc] = c01[r] * inv0 + bias1;
    o1[mloc]      = c10[r] * inv1 + bias0;
    o1[16 + mloc] = c11[r] * inv1 + bias1;
  }
}

// ---------------------------------------------------------------------------
extern "C" void kernel_launch(void* const* d_in, const int* in_sizes, int n_in,
                              void* d_out, int out_size, void* d_ws, size_t ws_size,
                              hipStream_t stream) {
  const float* h     = (const float*)d_in[0];
  const float* w     = (const float*)d_in[1];
  const float* a_src = (const float*)d_in[2];
  const float* a_dst = (const float*)d_in[3];
  const float* b     = (const float*)d_in[4];
  float* out = (float*)d_out;

  const size_t HP_BYTES = (size_t)N_HEAD * N_NODES * F_OUT * sizeof(unsigned short);
  const size_t SD_BYTES = (size_t)N_HEAD * N_NODES * sizeof(float);
  char* ws = (char*)d_ws;
  unsigned short* hp = (unsigned short*)ws;
  float* sArr = (float*)(ws + HP_BYTES);
  float* dArr = (float*)(ws + HP_BYTES + SD_BYTES);
  float* dmax = (float*)(ws + HP_BYTES + 2 * SD_BYTES);

  gat_prep<<<dim3(N_NODES / 8, N_HEAD), 256, 0, stream>>>(h, w, a_src, a_dst,
                                                          hp, sArr, dArr);
  gat_dmax<<<dim3(N_HEAD), 256, 0, stream>>>(dArr, dmax);
  gat_attn<<<dim3(N_NODES / 128, N_HEAD), 128, 0, stream>>>(sArr, dArr, hp,
                                                            dmax, b, out);
}
